// AdditiveAttention_29523605193363
// MI455X (gfx1250) — matrix operand order repaired and checked by reading, not compiled, not run on hardware
//
#include <hip/hip_runtime.h>
#include <math.h>

typedef __attribute__((ext_vector_type(16))) _Float16 v16h;
typedef __attribute__((ext_vector_type(8)))  float    v8f;
typedef __attribute__((ext_vector_type(2)))  float    v2f;

#if __has_builtin(__builtin_amdgcn_tanhf)
#define FAST_TANH(x) __builtin_amdgcn_tanhf(x)
#elif __has_builtin(__builtin_amdgcn_tanh_f32)
#define FAST_TANH(x) __builtin_amdgcn_tanh_f32(x)
#else
#define FAST_TANH(x) tanhf(x)
#endif

#if __has_builtin(__builtin_amdgcn_wmma_f32_16x16x4_f32)
#define HAVE_WMMA_F32X4 1
#endif

#define NB 2
#define NH 8
#define LQ 512
#define LK 512
#define DD 64
#define ROWS (NB * NH * LQ) /* 8192 */

// ---------------------------------------------------------------------------
// Kernel 1: Wq = Q@W1 + b1 ; Wk = K@W2 + b2   (fp32-exact via WMMA 16x16x4)
// blockIdx.y selects tensor (0=Q,1=K). 4 waves/block; wave tile = 16 rows x 64.
// ---------------------------------------------------------------------------
__global__ __launch_bounds__(128) void proj_kernel(
    const float* __restrict__ Q, const float* __restrict__ Km,
    const float* __restrict__ W1, const float* __restrict__ b1,
    const float* __restrict__ W2, const float* __restrict__ b2,
    float* __restrict__ Wq, float* __restrict__ Wk)
{
  const int which = blockIdx.y;
  const float* __restrict__ X    = which ? Km : Q;
  const float* __restrict__ W    = which ? W2 : W1;
  const float* __restrict__ bias = which ? b2 : b1;
  float* __restrict__ Y          = which ? Wk : Wq;

  // W transposed in LDS: Wt[c*66 + r] = W[r][c]; stride 66 keeps float2 reads
  // 8B-aligned and bank-conflict-free.
  __shared__ float Wt[64 * 66];
  __shared__ float bl[64];

  const int tid = threadIdx.x;
  for (int i = tid; i < 64 * 64; i += 128) {
    const int r = i >> 6, c = i & 63;
    Wt[c * 66 + r] = W[i];
  }
  if (tid < 64) bl[tid] = bias[tid];
  __syncthreads();

#if defined(HAVE_WMMA_F32X4)
  const int wave = tid >> 5, lane = tid & 31;
  const int half = lane >> 4;          // 0: K pair {0,1}, 1: K pair {2,3}
  const int col  = lane & 15;
  const long rowBase = (long)blockIdx.x * 64 + wave * 16;
  const long rowAbs  = rowBase + col;  // A-matrix row held by this lane

  v8f acc[4] = {};
#pragma unroll
  for (int kk = 0; kk < 64; kk += 4) {
    // A 16x4 fp32: lane<16 holds (row, K=kk..kk+1), lane>=16 holds K=kk+2..kk+3
    v2f a;
    { const float2 t2 = *(const float2*)(X + rowAbs * 64 + kk + 2 * half);
      a.x = t2.x; a.y = t2.y; }
    const int r0 = kk + 2 * half;      // B rows for this lane's half
#pragma unroll
    for (int t = 0; t < 4; ++t) {      // 4 column tiles of 16
      v2f b;
      const float2 t2 = *(const float2*)(&Wt[(t * 16 + col) * 66 + r0]);
      b.x = t2.x; b.y = t2.y;
      acc[t] = __builtin_amdgcn_wmma_f32_16x16x4_f32(
          false, a, false, b, (short)0, acc[t], false, false);
    }
  }
  // C layout: lane -> col (lane&15), rows 8*half + i
#pragma unroll
  for (int t = 0; t < 4; ++t) {
    const float bb = bl[t * 16 + col];
    const long rw  = rowBase + 8 * half;
#pragma unroll
    for (int i = 0; i < 8; ++i)
      Y[(rw + i) * 64 + t * 16 + col] = acc[t][i] + bb;
  }
#else
  // VALU fallback (fp32 dot)
  for (int o = tid; o < 64 * 64; o += 128) {
    const long r = (long)blockIdx.x * 64 + (o >> 6);
    const int  c = o & 63;
    float s = bl[c];
    const float* xr = X + r * 64;
#pragma unroll
    for (int d = 0; d < 64; ++d) s += xr[d] * Wt[c * 66 + d];
    Y[r * 64 + c] = s;
  }
#endif
}

// ---------------------------------------------------------------------------
// Kernel 2: scores[b,h,q,k] = sum_d tanh(Wq[q,d]+Wk[k,d]) * V[d] + bV
// Block = 4 waves, tile = 16q x 64k (16x16 per wave). Per wave: 32 x
// v_wmma_f32_16x16x32_f16, B = V in column k' (column-select reduction),
// so the Sum_d*V runs on the matrix pipe while VALU/TRANS do add+tanh.
// ---------------------------------------------------------------------------
__global__ __launch_bounds__(128) void scores_kernel(
    const float* __restrict__ Wq, const float* __restrict__ Wk,
    const float* __restrict__ V,  const float* __restrict__ bV,
    float* __restrict__ out)
{
  __shared__ float wqLds[16 * 68];   // padded stride 68: conflict-free, 16B rows
  __shared__ float wkLds[64 * 64];   // broadcast reads: no padding needed
  __shared__ float vLds[64];

  const int tid = threadIdx.x;
  const int kt = blockIdx.x & 7;
  const int qt = (blockIdx.x >> 3) & 31;
  const int bh = blockIdx.x >> 8;
  const int qbase = qt << 4, kbase = kt << 6;
  const float* __restrict__ WqT = Wq + ((long)bh * LQ + qbase) * DD;
  const float* __restrict__ WkT = Wk + ((long)bh * LK + kbase) * DD;

  for (int i = tid; i < 256; i += 128) {        // 16x64 Wq tile
    const int r = i >> 4, c4 = (i & 15) << 2;
    *(float4*)&wqLds[r * 68 + c4] = *(const float4*)(WqT + r * 64 + c4);
  }
  for (int i = tid; i < 1024; i += 128) {       // 64x64 Wk tile
    const int r = i >> 4, c4 = (i & 15) << 2;
    *(float4*)&wkLds[r * 64 + c4] = *(const float4*)(WkT + r * 64 + c4);
  }
  if (tid < 64) vLds[tid] = V[tid];
  __syncthreads();

  const int wave = tid >> 5, lane = tid & 31;
  const int half = lane >> 4, col = lane & 15;

  // A-operand f16 16x32 slot map (ISA 7.12.2): element i of lane (half,row=col)
  //   -> d = chunk*32 + 16*(i>>3) + 8*half + (i&7)
  // Cache this lane's 32 relevant Wq values in VGPRs (reused for all 16 k').
  const float* wqrow = &wqLds[col * 68 + half * 8];
  float wqr[32];
#pragma unroll
  for (int chunk = 0; chunk < 2; ++chunk)
#pragma unroll
    for (int g = 0; g < 2; ++g)
#pragma unroll
      for (int j = 0; j < 8; ++j)
        wqr[chunk * 16 + g * 8 + j] = wqrow[chunk * 32 + g * 16 + j];

  // B-operand f16 32x16 slot map: lane holds column (lane&15), rows 16*(i>>3)+8*half+(i&7).
  v16h vh[2];
#pragma unroll
  for (int chunk = 0; chunk < 2; ++chunk)
#pragma unroll
    for (int i = 0; i < 16; ++i)
      vh[chunk][i] = (_Float16)vLds[chunk * 32 + 16 * (i >> 3) + 8 * half + (i & 7)];

  v8f c = {};
  for (int kp = 0; kp < 16; ++kp) {
    const float* wkrow = &wkLds[(wave * 16 + kp) * 64 + half * 8];
    const bool on = (col == kp);   // this lane's B column carries V for k'==kp
#pragma unroll
    for (int chunk = 0; chunk < 2; ++chunk) {
      v16h a;
#pragma unroll
      for (int g = 0; g < 2; ++g)
#pragma unroll
        for (int j = 0; j < 8; ++j) {
          const float x = wqr[chunk * 16 + g * 8 + j] + wkrow[chunk * 32 + g * 16 + j];
          a[g * 8 + j] = (_Float16)FAST_TANH(x);
        }
      v16h b;
#pragma unroll
      for (int i = 0; i < 16; ++i)
        b[i] = on ? vh[chunk][i] : (_Float16)0.0f;   // cndmask, EXEC stays full
      c = __builtin_amdgcn_wmma_f32_16x16x32_f16(
          false, a, false, b, (short)0, c, false, false);
    }
  }

  // C layout: lane -> k col (lane&15), q rows 8*half + i
  const float biasV = bV[0];
  const int  kIdx = kbase + (wave << 4) + col;
  const long outBase = (((long)bh * LQ) + qbase + 8 * half) * (long)LK + kIdx;
#pragma unroll
  for (int i = 0; i < 8; ++i)
    out[outBase + (long)i * LK] = c[i] + biasV;
}

// ---------------------------------------------------------------------------
extern "C" void kernel_launch(void* const* d_in, const int* in_sizes, int n_in,
                              void* d_out, int out_size, void* d_ws, size_t ws_size,
                              hipStream_t stream) {
  (void)in_sizes; (void)n_in; (void)out_size; (void)ws_size;
  const float* Q  = (const float*)d_in[0];
  const float* K  = (const float*)d_in[1];
  const float* W1 = (const float*)d_in[2];
  const float* b1 = (const float*)d_in[3];
  const float* W2 = (const float*)d_in[4];
  const float* b2 = (const float*)d_in[5];
  const float* V  = (const float*)d_in[6];
  const float* bV = (const float*)d_in[7];

  float* Wq = (float*)d_ws;                  // 8192*64 f32 = 2 MB
  float* Wk = Wq + (size_t)ROWS * DD;        // next 2 MB
  float* out = (float*)d_out;

  proj_kernel<<<dim3(ROWS / 64, 2, 1), 128, 0, stream>>>(Q, K, W1, b1, W2, b2, Wq, Wk);
  scores_kernel<<<dim3((NB * NH) * (LQ / 16) * (LK / 64), 1, 1), 128, 0, stream>>>(
      Wq, Wk, V, bV, out);
}
